// Decoder_566935683216
// MI455X (gfx1250) — matrix operand order repaired and checked by reading, not compiled, not run on hardware
//
#include <hip/hip_runtime.h>
#include <hip/hip_bf16.h>

#define B_  32
#define L_  64
#define T_  32
#define H_  512
#define E_  300
#define EP_ 320     // E padded to multiple of 32 for WMMA K
#define V_  32000

typedef __attribute__((ext_vector_type(16))) __bf16       v16bf;
typedef __attribute__((ext_vector_type(8)))  float        v8f;
typedef __attribute__((ext_vector_type(4)))  unsigned int u32x4;

union FragBF {
  v16bf v;
  u32x4 q[2];
};

// A-matrix 16x32 bf16 fragment (ISA 7.12.2): lane<16 holds row M=lane, K={0..7,16..23};
// lane>=16 holds row M=lane-16, K={8..15,24..31}.  rowp points at A[row][0].
__device__ __forceinline__ v16bf load_a_frag(const __hip_bfloat16* rowp, int k, int lane) {
  const int khalf = (lane >> 4) << 3;   // 0 or 8
  FragBF f;
  f.q[0] = *reinterpret_cast<const u32x4*>(rowp + k + khalf);
  f.q[1] = *reinterpret_cast<const u32x4*>(rowp + k + 16 + khalf);
  return f.v;
}

// B-matrix 32x16 bf16 fragment: lane<16 holds col N=lane, K={0..7,16..23}; lane>=16 holds
// col N=lane-16, K={8..15,24..31} (same K order as A).  colp points at Bt[col][0]
// where Bt is the N x K (pre-transposed) weight.
__device__ __forceinline__ v16bf load_b_frag(const __hip_bfloat16* colp, int k, int lane) {
  const int koff = (lane >> 4) << 3;    // 0 or 8
  FragBF f;
  f.q[0] = *reinterpret_cast<const u32x4*>(colp + k + koff);
  f.q[1] = *reinterpret_cast<const u32x4*>(colp + k + 16 + koff);
  return f.v;
}

#define WMMA_BF16(a, b, c) \
  __builtin_amdgcn_wmma_f32_16x16x32_bf16(false, (a), false, (b), (short)0, (c), false, false)

// C[M][N] = A[M][K] (bf16 row-major) x Bt[N][K] (bf16, weight pre-transposed) + bias[N]
// 256 threads = 8 waves, arranged WAVES_M x (8/WAVES_M).  Each wave owns a 32x32 quad
// of 16x16 tiles: 4 WMMAs per K-slice on 4 fragment loads (2 A + 2 B).  The steady-state
// loop is branchless: the next K-slice's fragments are loaded unconditionally before the
// current WMMAs issue (software pipeline, partial s_wait_loadcnt), and the final K-slice
// is peeled so no in-loop guard or register shuffling is needed.  Four independent
// accumulator chains co-execute through the bf16 WMMA->WMMA hazard window.
// Requires: M % (WAVES_M*32) == 0, N % ((8/WAVES_M)*32) == 0, K % 32 == 0, K >= 32.
template<int WAVES_M, bool NTSTORE>
__global__ void gemm_bf16_wmma(const __hip_bfloat16* __restrict__ A,
                               const __hip_bfloat16* __restrict__ Bt,
                               const float* __restrict__ bias,
                               float* __restrict__ C,
                               int M, int N, int K) {
  constexpr int WAVES_N = 8 / WAVES_M;
  const int lane = threadIdx.x & 31;
  const int wave = threadIdx.x >> 5;
  const int wm = wave / WAVES_N;
  const int wn = wave % WAVES_N;
  const int m0 = blockIdx.y * (WAVES_M * 32) + wm * 32;
  const int n0 = blockIdx.x * (WAVES_N * 32) + wn * 32;
  const int r = lane & 15;

  const __hip_bfloat16* arow0 = A + (size_t)(m0 + r) * K;
  const __hip_bfloat16* arow1 = arow0 + (size_t)16 * K;
  const __hip_bfloat16* bcol0 = Bt + (size_t)(n0 + r) * K;
  const __hip_bfloat16* bcol1 = bcol0 + (size_t)16 * K;

  v8f c00 = {}, c01 = {}, c10 = {}, c11 = {};

  // pipeline prologue: fragments for k = 0
  v16bf a0 = load_a_frag(arow0, 0, lane);
  v16bf a1 = load_a_frag(arow1, 0, lane);
  v16bf b0 = load_b_frag(bcol0, 0, lane);
  v16bf b1 = load_b_frag(bcol1, 0, lane);

  // steady state: branchless; loads for k+32 issue ahead of the WMMAs for k
  int k = 0;
  for (; k + 32 < K; k += 32) {
    const int kn = k + 32;
    __builtin_prefetch(arow0 + kn + 32, 0, 1);
    __builtin_prefetch(bcol0 + kn + 32, 0, 1);
    v16bf na0 = load_a_frag(arow0, kn, lane);
    v16bf na1 = load_a_frag(arow1, kn, lane);
    v16bf nb0 = load_b_frag(bcol0, kn, lane);
    v16bf nb1 = load_b_frag(bcol1, kn, lane);
    c00 = WMMA_BF16(a0, b0, c00);
    c01 = WMMA_BF16(a0, b1, c01);
    c10 = WMMA_BF16(a1, b0, c10);
    c11 = WMMA_BF16(a1, b1, c11);
    a0 = na0; a1 = na1; b0 = nb0; b1 = nb1;
  }
  // peeled final K-slice
  c00 = WMMA_BF16(a0, b0, c00);
  c01 = WMMA_BF16(a0, b1, c01);
  c10 = WMMA_BF16(a1, b0, c10);
  c11 = WMMA_BF16(a1, b1, c11);

  const int col0 = n0 + r;                        // D: N = lane&15 within tile
  const int col1 = col0 + 16;
  const float bv0 = bias ? bias[col0] : 0.0f;
  const float bv1 = bias ? bias[col1] : 0.0f;
  const int rbase = m0 + ((lane >> 4) << 3);      // lane>=16 -> rows 8..15 of tile
#pragma unroll
  for (int i = 0; i < 8; ++i) {
    float o00 = c00[i] + bv0, o01 = c01[i] + bv1;
    float o10 = c10[i] + bv0, o11 = c11[i] + bv1;
    float* p0 = C + (size_t)(rbase + i) * N;
    float* p1 = C + (size_t)(rbase + 16 + i) * N;
    if (NTSTORE) {
      // streaming output: never re-read, keep it out of L2 so W_out stays resident
      __builtin_nontemporal_store(o00, p0 + col0);
      __builtin_nontemporal_store(o01, p0 + col1);
      __builtin_nontemporal_store(o10, p1 + col0);
      __builtin_nontemporal_store(o11, p1 + col1);
    } else {
      p0[col0] = o00; p0[col1] = o01;
      p1[col0] = o10; p1[col1] = o11;
    }
  }
}

// ---- conversion / staging kernels -------------------------------------------------

__global__ void cvt_bf16(const float* __restrict__ src, __hip_bfloat16* __restrict__ dst, int n) {
  int i = blockIdx.x * blockDim.x + threadIdx.x;
  if (i < n) dst[i] = __float2bfloat16(src[i]);
}

// src [Ksrc][N] f32 row-major  ->  dst [N][Kdst] bf16 (zero-padded K)
__global__ void cvt_transpose(const float* __restrict__ src, __hip_bfloat16* __restrict__ dst,
                              int Ksrc, int N, int Kdst) {
  int i = blockIdx.x * blockDim.x + threadIdx.x;
  if (i >= N * Kdst) return;
  int n = i / Kdst, k = i - n * Kdst;
  float v = (k < Ksrc) ? src[(size_t)k * N + n] : 0.0f;
  dst[(size_t)n * Kdst + k] = __float2bfloat16(v);
}

// gathered, padded embeddings: dst[(b*T+t)][0..EP) bf16
__global__ void gather_emb(const int* __restrict__ words, const float* __restrict__ embed,
                           __hip_bfloat16* __restrict__ dst) {
  int i = blockIdx.x * blockDim.x + threadIdx.x;    // B_*T_*EP_
  if (i >= B_ * T_ * EP_) return;
  int row = i / EP_, k = i - row * EP_;
  int w = words[row];
  float v = (k < E_) ? embed[(size_t)w * E_ + k] : 0.0f;
  dst[i] = __float2bfloat16(v);
}

__global__ void build_bias_azr(const float* __restrict__ b_a, const float* __restrict__ b_hzr,
                               float* __restrict__ out) {
  int i = blockIdx.x * blockDim.x + threadIdx.x;    // 3H
  if (i >= 3 * H_) return;
  out[i] = (i < H_) ? b_a[i] : b_hzr[i - H_];
}

__global__ void init_s(const float* __restrict__ prev_s, float* __restrict__ s,
                       __hip_bfloat16* __restrict__ sA) {
  int i = blockIdx.x * blockDim.x + threadIdx.x;    // B_*H_
  if (i >= B_ * H_) return;
  float v = prev_s[i];
  s[i] = v;
  sA[i] = __float2bfloat16(v);
}

// ---- per-step kernels -------------------------------------------------------------

// One block per batch row. sW[b][0..H) = out_s.  Computes Bahdanau attention and
// context vector; emits ctx in f32 and bf16 (next GEMM operand).
__global__ void attn_kernel(const float* __restrict__ sW,
                            const float* __restrict__ Uh,      // [B*L][H]
                            const float* __restrict__ enc_h,   // [B][L][2H]
                            const float* __restrict__ v_w,
                            const float* __restrict__ v_b,
                            float* __restrict__ ctx,           // [B][2H]
                            __hip_bfloat16* __restrict__ ctxA) {
  const int b = blockIdx.x;
  const int tid = threadIdx.x;                      // 256
  __shared__ float s_out[H_];
  __shared__ float sc[L_];
  __shared__ float red[256];
  __shared__ float ssum;

  for (int h = tid; h < H_; h += 256) s_out[h] = sW[b * (3 * H_) + h];
  __syncthreads();

  // 4 threads per encoder position
  const int l = tid >> 2;
  const int sub = tid & 3;
  const float* uh = Uh + (size_t)(b * L_ + l) * H_;
  float p = 0.0f;
  for (int h = sub; h < H_; h += 4) p += v_w[h] * tanhf(s_out[h] + uh[h]);
  red[tid] = p;
  __syncthreads();
  if (sub == 0) sc[l] = red[tid] + red[tid + 1] + red[tid + 2] + red[tid + 3] + v_b[0];
  __syncthreads();

  if (tid == 0) {
    float mx = sc[0];
    for (int i = 1; i < L_; ++i) mx = fmaxf(mx, sc[i]);
    float sm = 0.0f;
    for (int i = 0; i < L_; ++i) { sc[i] = expf(sc[i] - mx); sm += sc[i]; }
    ssum = sm;
  }
  __syncthreads();
  const float inv = 1.0f / ssum;

  for (int d = tid; d < 2 * H_; d += 256) {
    const float* eh = enc_h + (size_t)b * L_ * (2 * H_) + d;
    float acc = 0.0f;
    for (int l2 = 0; l2 < L_; ++l2) acc += sc[l2] * eh[(size_t)l2 * (2 * H_)];
    acc *= inv;
    ctx[b * (2 * H_) + d] = acc;
    ctxA[b * (2 * H_) + d] = __float2bfloat16(acc);
  }
}

// z, r gates; emits z (f32) and r*s (bf16) for the W_hid_s GEMM.
__global__ void gate_kernel(const float* __restrict__ GE,    // [B*T][3H]
                            const float* __restrict__ sW,    // [B][3H] (out_s | gh)
                            const float* __restrict__ gc,    // [B][3H]
                            const float* __restrict__ s,     // [B][H]
                            float* __restrict__ zbuf,        // [B][H]
                            __hip_bfloat16* __restrict__ rsA,
                            int t) {
  int i = blockIdx.x * blockDim.x + threadIdx.x;   // B_*H_
  if (i >= B_ * H_) return;
  int b = i >> 9, j = i & (H_ - 1);
  size_t gerow = (size_t)(b * T_ + t) * (3 * H_);
  size_t swrow = (size_t)b * (3 * H_);
  float zv = GE[gerow + j]        + sW[swrow + H_ + j]      + gc[swrow + j];
  float rv = GE[gerow + H_ + j]   + sW[swrow + 2 * H_ + j]  + gc[swrow + H_ + j];
  zv = 1.0f / (1.0f + expf(-zv));
  rv = 1.0f / (1.0f + expf(-rv));
  zbuf[i] = zv;
  rsA[i] = __float2bfloat16(rv * s[i]);
}

// s_tilde + state update; emits new state in f32 and bf16, and appends to s_seq.
__global__ void update_kernel(const float* __restrict__ GE,
                              const float* __restrict__ gc,
                              const float* __restrict__ hs,   // (r*s)@W_hid_s + b_hid_s
                              const float* __restrict__ zbuf,
                              float* __restrict__ s,
                              __hip_bfloat16* __restrict__ sA,
                              float* __restrict__ s_seq,
                              __hip_bfloat16* __restrict__ s_seqA,
                              int t) {
  int i = blockIdx.x * blockDim.x + threadIdx.x;   // B_*H_
  if (i >= B_ * H_) return;
  int b = i >> 9, j = i & (H_ - 1);
  size_t gerow = (size_t)(b * T_ + t) * (3 * H_);
  float st = tanhf(GE[gerow + 2 * H_ + j] + hs[i] + gc[(size_t)b * (3 * H_) + 2 * H_ + j]);
  float z = zbuf[i];
  float sn = (1.0f - z) * s[i] + z * st;
  s[i] = sn;
  sA[i] = __float2bfloat16(sn);
  size_t row = (size_t)(b * T_ + t) * H_ + j;
  s_seq[row]  = sn;
  s_seqA[row] = __float2bfloat16(sn);
}

// -----------------------------------------------------------------------------------

#define CDIV(a, b) (((a) + (b) - 1) / (b))

extern "C" void kernel_launch(void* const* d_in, const int* in_sizes, int n_in,
                              void* d_out, int out_size, void* d_ws, size_t ws_size,
                              hipStream_t stream) {
  (void)in_sizes; (void)n_in; (void)out_size; (void)ws_size;

  const float* enc_h  = (const float*)d_in[0];
  const float* prev_s = (const float*)d_in[1];
  const int*   tgt    = (const int*)d_in[2];
  const float* embed  = (const float*)d_in[3];
  const float* W_a    = (const float*)d_in[4];
  const float* b_a    = (const float*)d_in[5];
  const float* U_a    = (const float*)d_in[6];
  const float* b_Ua   = (const float*)d_in[7];
  const float* v_w    = (const float*)d_in[8];
  const float* v_b    = (const float*)d_in[9];
  const float* W_emb  = (const float*)d_in[10];
  const float* b_emb  = (const float*)d_in[11];
  const float* W_hzr  = (const float*)d_in[12];
  const float* b_hzr  = (const float*)d_in[13];
  const float* W_hs   = (const float*)d_in[14];
  const float* b_hs   = (const float*)d_in[15];
  const float* W_ctx  = (const float*)d_in[16];
  const float* b_ctx  = (const float*)d_in[17];
  const float* W_out  = (const float*)d_in[18];
  const float* b_out  = (const float*)d_in[19];

  char* ws = (char*)d_ws;
  size_t off = 0;
  auto alloc = [&](size_t bytes) -> char* {
    char* p = ws + off;
    off = (off + bytes + 255) & ~(size_t)255;
    return p;
  };
  typedef __hip_bfloat16 bf;

  bf*    encA     = (bf*)   alloc((size_t)B_ * L_ * 2 * H_ * 2);       // [2048][1024] bf16
  bf*    UaT      = (bf*)   alloc((size_t)H_ * 2 * H_ * 2);            // [512][1024]
  float* Uh       = (float*)alloc((size_t)B_ * L_ * H_ * 4);           // [2048][512]
  bf*    embA     = (bf*)   alloc((size_t)B_ * T_ * EP_ * 2);          // [1024][320]
  bf*    WembT    = (bf*)   alloc((size_t)3 * H_ * EP_ * 2);           // [1536][320]
  float* GE       = (float*)alloc((size_t)B_ * T_ * 3 * H_ * 4);       // [1024][1536]
  bf*    WazrT    = (bf*)   alloc((size_t)3 * H_ * H_ * 2);            // [1536][512]
  bf*    WctxT    = (bf*)   alloc((size_t)3 * H_ * 2 * H_ * 2);        // [1536][1024]
  bf*    WhsT     = (bf*)   alloc((size_t)H_ * H_ * 2);                // [512][512]
  bf*    WoutT    = (bf*)   alloc((size_t)V_ * H_ * 2);                // [32000][512]
  float* bias_azr = (float*)alloc((size_t)3 * H_ * 4);
  float* s_cur    = (float*)alloc((size_t)B_ * H_ * 4);
  bf*    sA       = (bf*)   alloc((size_t)B_ * H_ * 2);
  float* sW       = (float*)alloc((size_t)B_ * 3 * H_ * 4);            // [32][1536]
  float* ctx      = (float*)alloc((size_t)B_ * 2 * H_ * 4);
  bf*    ctxA     = (bf*)   alloc((size_t)B_ * 2 * H_ * 2);
  float* gc       = (float*)alloc((size_t)B_ * 3 * H_ * 4);
  float* zbuf     = (float*)alloc((size_t)B_ * H_ * 4);
  bf*    rsA      = (bf*)   alloc((size_t)B_ * H_ * 2);
  float* hs       = (float*)alloc((size_t)B_ * H_ * 4);
  float* s_seq    = (float*)alloc((size_t)B_ * T_ * H_ * 4);
  bf*    s_seqA   = (bf*)   alloc((size_t)B_ * T_ * H_ * 2);
  float* logits   = (float*)d_out;                                     // [1024][32000]

  const int TB = 256;

  // ---- one-time staging (recomputed every call; deterministic) ----
  cvt_bf16<<<CDIV(B_ * L_ * 2 * H_, TB), TB, 0, stream>>>(enc_h, encA, B_ * L_ * 2 * H_);
  cvt_transpose<<<CDIV(H_ * 2 * H_, TB), TB, 0, stream>>>(U_a, UaT, 2 * H_, H_, 2 * H_);
  cvt_transpose<<<CDIV(3 * H_ * EP_, TB), TB, 0, stream>>>(W_emb, WembT, E_, 3 * H_, EP_);
  cvt_transpose<<<CDIV(H_ * H_, TB), TB, 0, stream>>>(W_a, WazrT, H_, H_, H_);
  cvt_transpose<<<CDIV(2 * H_ * H_, TB), TB, 0, stream>>>(W_hzr, WazrT + (size_t)H_ * H_, H_, 2 * H_, H_);
  cvt_transpose<<<CDIV(3 * H_ * 2 * H_, TB), TB, 0, stream>>>(W_ctx, WctxT, 2 * H_, 3 * H_, 2 * H_);
  cvt_transpose<<<CDIV(H_ * H_, TB), TB, 0, stream>>>(W_hs, WhsT, H_, H_, H_);
  cvt_transpose<<<CDIV((size_t)V_ * H_, (size_t)TB), TB, 0, stream>>>(W_out, WoutT, H_, V_, H_);
  gather_emb<<<CDIV(B_ * T_ * EP_, TB), TB, 0, stream>>>(tgt, embed, embA);
  build_bias_azr<<<CDIV(3 * H_, TB), TB, 0, stream>>>(b_a, b_hzr, bias_azr);
  init_s<<<CDIV(B_ * H_, TB), TB, 0, stream>>>(prev_s, s_cur, sA);

  // Uh = enc_h @ U_a + b_Ua : M=2048 N=512 K=1024  (block tile 64x128)
  gemm_bf16_wmma<2, false><<<dim3(H_ / 128, (B_ * L_) / 64), TB, 0, stream>>>(
      encA, UaT, b_Ua, Uh, B_ * L_, H_, 2 * H_);
  // GE = emb @ W_emb + b_emb : M=1024 N=1536 K=320 (hoisted out of the loop)
  gemm_bf16_wmma<2, false><<<dim3((3 * H_) / 128, (B_ * T_) / 64), TB, 0, stream>>>(
      embA, WembT, b_emb, GE, B_ * T_, 3 * H_, EP_);

  // ---- sequential decode loop ----
  for (int t = 0; t < T_; ++t) {
    // sW = s @ [W_a | W_hid_zr] + [b_a | b_hid_zr] : M=32 N=1536 K=512 (tile 32x256)
    gemm_bf16_wmma<1, false><<<dim3((3 * H_) / 256, 1), TB, 0, stream>>>(
        sA, WazrT, bias_azr, sW, B_, 3 * H_, H_);
    attn_kernel<<<B_, TB, 0, stream>>>(sW, Uh, enc_h, v_w, v_b, ctx, ctxA);
    // gc = ctx @ W_ctx + b_ctx : M=32 N=1536 K=1024
    gemm_bf16_wmma<1, false><<<dim3((3 * H_) / 256, 1), TB, 0, stream>>>(
        ctxA, WctxT, b_ctx, gc, B_, 3 * H_, 2 * H_);
    gate_kernel<<<CDIV(B_ * H_, TB), TB, 0, stream>>>(GE, sW, gc, s_cur, zbuf, rsA, t);
    // hs = (r*s) @ W_hid_s + b_hid_s : M=32 N=512 K=512
    gemm_bf16_wmma<1, false><<<dim3(H_ / 256, 1), TB, 0, stream>>>(
        rsA, WhsT, b_hs, hs, B_, H_, H_);
    update_kernel<<<CDIV(B_ * H_, TB), TB, 0, stream>>>(
        GE, gc, hs, zbuf, s_cur, sA, s_seq, s_seqA, t);
  }

  // logits = s_seq @ W_out + b_out : M=1024 N=32000 K=512  — NT stores for the
  // 131 MB streaming output so W_out (32 MB) stays resident in the 192 MB L2.
  gemm_bf16_wmma<2, true><<<dim3(V_ / 128, (B_ * T_) / 64), TB, 0, stream>>>(
      s_seqA, WoutT, b_out, logits, B_ * T_, V_, H_);
}
